// MultiQueryAttention_28656021799163
// MI455X (gfx1250) — matrix operand order repaired and checked by reading, not compiled, not run on hardware
//
#include <hip/hip_runtime.h>

// ---------------------------------------------------------------------------
// Multi-query attention, CDNA5 (gfx1250), wave32 + WMMA f16->f32.
//
// Pipeline:
//   1) x (f32) -> xh (f16);  W* (f32, [K,N]) -> W*t (f16, [N,K])  (transposed)
//   2) Qh  = xh @ Wq + bq              (WMMA GEMM, f16 out)
//      Kh  = xh @ Wk + bk              (WMMA GEMM, f16 out)
//      Vt  = (xh @ Wv + bv)^T          (WMMA GEMM, transposed f16 out [dk, B*T])
//   3) flash MQA: per 16-query tile, stream keys 32 at a time,
//      online softmax, P relayout via per-wave LDS, P@V accumulation.
//   4) d_out = AO @ Wo + bo            (WMMA GEMM, f32 out)
//
// GEMM stages its B tile through LDS with gfx1250 async copies
// (GLOBAL_LOAD_ASYNC_TO_LDS_B128 + s_wait_asynccnt), double buffered,
// shared by all 8 waves of a 128-row M block.
// ---------------------------------------------------------------------------

#define B_SZ   2
#define T_SZ   2048
#define D_SZ   2048
#define H_SZ   16
#define DK_SZ  128
#define BT_SZ  (B_SZ * T_SZ)   // 4096 rows

typedef __attribute__((ext_vector_type(16))) _Float16 v16h;
typedef __attribute__((ext_vector_type(8)))  float    v8f;
typedef int v4i_gld __attribute__((vector_size(16)));

union Frag16 {
    uint4 u2[2];
    v16h  h;
};

// A-matrix 16x32 f16 fragment (row-major source, row pointer already at k):
// lane r (=lane&15) holds row m0+r; K = 16*hf .. 16*hf+15 contiguous.
__device__ __forceinline__ v16h load_a_frag(const _Float16* rowk, int hf) {
    Frag16 f;
    const uint4* p = (const uint4*)(rowk + 16 * hf);
    f.u2[0] = p[0];   // elements [16h, 16h+8)
    f.u2[1] = p[1];   // elements [16h+8, 16h+16)
    return f.h;
}

// B-matrix 32x16 f16 fragment from a [N,K]-style row (row = column n,
// pointer already at k): lane holds column n, K = 16*hf .. 16*hf+15 contiguous.
// Works for both global and LDS-resident rows (addrspace inferred on inline).
__device__ __forceinline__ v16h load_b_frag(const _Float16* rowk, int hf) {
    Frag16 f;
    const uint4* p = (const uint4*)(rowk + 16 * hf);
    f.u2[0] = p[0];
    f.u2[1] = p[1];
    return f.h;
}

__device__ __forceinline__ v8f wmma_f16(v16h a, v16h b, v8f c) {
    return __builtin_amdgcn_wmma_f32_16x16x32_f16(false, a, false, b,
                                                  (short)0, c, false, false);
}

__device__ __forceinline__ float half_reduce_max(float v) {
    v = fmaxf(v, __shfl_xor(v, 1, 32));
    v = fmaxf(v, __shfl_xor(v, 2, 32));
    v = fmaxf(v, __shfl_xor(v, 4, 32));
    v = fmaxf(v, __shfl_xor(v, 8, 32));
    return v;
}
__device__ __forceinline__ float half_reduce_sum(float v) {
    v += __shfl_xor(v, 1, 32);
    v += __shfl_xor(v, 2, 32);
    v += __shfl_xor(v, 4, 32);
    v += __shfl_xor(v, 8, 32);
    return v;
}

// ---------------------------------------------------------------------------
// gfx1250 async copy global -> LDS (one 16-byte chunk per calling lane),
// tracked with ASYNCcnt.
// Builtin signature (probed from this toolchain's diagnostic):
//   (v4i addrspace(1)* src, v4i addrspace(3)* dst, int offset, int cpol)
// ---------------------------------------------------------------------------
__device__ __forceinline__ void async_copy_b128(const _Float16* g, _Float16* l) {
#if __has_builtin(__builtin_amdgcn_global_load_async_to_lds_b128)
    __builtin_amdgcn_global_load_async_to_lds_b128(
        (__attribute__((address_space(1))) v4i_gld*)g,
        (__attribute__((address_space(3))) v4i_gld*)l, 0, 0);
#else
    unsigned lofs =
        (unsigned)(size_t)(__attribute__((address_space(3))) void*)l;
    unsigned long long ga = (unsigned long long)g;
    asm volatile("global_load_async_to_lds_b128 %0, %1, off"
                 :: "v"(lofs), "v"(ga) : "memory");
#endif
}

__device__ __forceinline__ void wait_async0() {
#if __has_builtin(__builtin_amdgcn_s_wait_asynccnt)
    __builtin_amdgcn_s_wait_asynccnt(0);
#else
    asm volatile("s_wait_asynccnt 0" ::: "memory");
#endif
}

// ---------------------------------------------------------------------------
// Converters
// ---------------------------------------------------------------------------
__global__ void cvt_f32_to_f16(const float* __restrict__ in,
                               _Float16* __restrict__ out, int n) {
    int i = blockIdx.x * blockDim.x + threadIdx.x;
    if (i < n) out[i] = (_Float16)in[i];
}

// in: [K,N] f32 (row-major), out: [N,K] f16 (row-major)
__global__ void cvt_transpose_f32_to_f16(const float* __restrict__ in,
                                         _Float16* __restrict__ out,
                                         int K, int N) {
    int i = blockIdx.x * blockDim.x + threadIdx.x;
    if (i < N * K) {
        int n = i / K;
        int k = i - n * K;
        out[i] = (_Float16)in[(size_t)k * N + n];
    }
}

// ---------------------------------------------------------------------------
// WMMA GEMM:  C[M,N] = A[M,K] @ Bt[N,K]^T + bias[N]
// Block = 8 waves = 128 M rows x one 64-col N strip.
// B tile (64 cols x 32 k = 4KB f16) staged through LDS with async copies,
// double buffered; every wave computes a 16x64 strip from the shared stage.
// Requires M % 128 == 0, N % 64 == 0, K % 32 == 0, exact grid.
// ---------------------------------------------------------------------------
template <bool OUT_F32, bool TRANS_OUT>
__global__ void __launch_bounds__(256)
gemm_wmma_f16(const _Float16* __restrict__ A,
              const _Float16* __restrict__ Bt,
              const float*    __restrict__ bias,
              void*           __restrict__ Cout,
              int M, int N, int K) {
    __shared__ __align__(16) _Float16 bstage[2][64][32];

    const int lane = threadIdx.x & 31;
    const int wv   = threadIdx.x >> 5;

    const int ntGroups = N >> 6;                        // 64-wide strips
    const int mBlk = blockIdx.x / ntGroups;
    const int nt0  = (blockIdx.x - mBlk * ntGroups) << 2;  // in 16-col tiles

    const int r  = lane & 15;
    const int hf = lane >> 4;
    const int m0 = (mBlk * 8 + wv) << 4;

    const _Float16* arow = A + (size_t)(m0 + r) * K;

    // Each of the 256 threads owns one 16-byte chunk of the 64x32 B stage:
    // row = tid>>2 (B column nt0*16+row), chunk = tid&3 (k offsets chunk*8..+8)
    const int tid  = threadIdx.x;
    const int srow = tid >> 2;
    const int schk = tid & 3;
    const _Float16* gsrc = Bt + (size_t)(nt0 * 16 + srow) * K + schk * 8;
    _Float16* ldst[2] = { &bstage[0][srow][schk * 8],
                          &bstage[1][srow][schk * 8] };

    v8f acc0 = {}, acc1 = {}, acc2 = {}, acc3 = {};

    // Prologue: stage k=0 into buffer 0.
    async_copy_b128(gsrc, ldst[0]);
    wait_async0();
    __syncthreads();

    const int nsteps = K >> 5;
    for (int s = 0; s < nsteps; ++s) {
        const int k = s << 5;
        if (s + 1 < nsteps)
            async_copy_b128(gsrc + k + 32, ldst[(s + 1) & 1]);

        v16h a = load_a_frag(arow + k, hf);
        const _Float16 (*bb)[32] = bstage[s & 1];
        acc0 = wmma_f16(a, load_b_frag(&bb[ 0 + r][0], hf), acc0);
        acc1 = wmma_f16(a, load_b_frag(&bb[16 + r][0], hf), acc1);
        acc2 = wmma_f16(a, load_b_frag(&bb[32 + r][0], hf), acc2);
        acc3 = wmma_f16(a, load_b_frag(&bb[48 + r][0], hf), acc3);

        wait_async0();       // next buffer staged
        __syncthreads();     // all waves done reading current buffer
    }

    v8f acc[4] = {acc0, acc1, acc2, acc3};
#pragma unroll
    for (int t = 0; t < 4; ++t) {
        const int col = (nt0 + t) * 16 + r;
        const float bb = bias[col];
#pragma unroll
        for (int i = 0; i < 8; ++i) {
            const int row = m0 + i + 8 * hf;
            const float v = acc[t][i] + bb;
            if (OUT_F32) {
                ((float*)Cout)[(size_t)row * N + col] = v;
            } else if (TRANS_OUT) {
                ((_Float16*)Cout)[(size_t)col * M + row] = (_Float16)v;
            } else {
                ((_Float16*)Cout)[(size_t)row * N + col] = (_Float16)v;
            }
        }
    }
}

// ---------------------------------------------------------------------------
// Flash multi-query attention.
//  Qh : [B*T, D] f16   (head h uses cols h*128..h*128+127)
//  Kh : [B*T, 128] f16 (shared across heads)
//  Vt : [128, B*T] f16 (transposed V, shared across heads)
//  AO : [B*T, D] f16
// Block = 4 waves; each wave owns a 16-query tile; grid = B*H*(T/64).
// ---------------------------------------------------------------------------
__global__ void __launch_bounds__(128)
flash_mqa(const _Float16* __restrict__ Qh,
          const _Float16* __restrict__ Kh,
          const _Float16* __restrict__ Vt,
          _Float16*       __restrict__ AO) {
    __shared__ __align__(16) _Float16 pbuf[4][16][32];

    const int lane = threadIdx.x & 31;
    const int wv   = threadIdx.x >> 5;
    const int r    = lane & 15;
    const int hf   = lane >> 4;

    const int qt = blockIdx.x % (T_SZ / 64);
    const int h  = (blockIdx.x / (T_SZ / 64)) % H_SZ;
    const int b  = blockIdx.x / ((T_SZ / 64) * H_SZ);
    const int q0 = qt * 64 + wv * 16;

    // Load the full 16x128 Q tile as 4 A fragments (kept in registers).
    const _Float16* qbase = Qh + (size_t)(b * T_SZ + q0 + r) * D_SZ + h * DK_SZ;
    v16h qa[4];
#pragma unroll
    for (int c = 0; c < 4; ++c) qa[c] = load_a_frag(qbase + 32 * c, hf);

    v8f o[8] = {};
    float mi[8], li[8];
#pragma unroll
    for (int i = 0; i < 8; ++i) { mi[i] = -INFINITY; li[i] = 0.0f; }

    const float scale = 0.08838834764831845f;  // 1/sqrt(128)

    for (int kk = 0; kk < T_SZ; kk += 32) {
        // ---- S = Q @ K^T for 32 keys (two 16-key column tiles) ----
        v8f s0 = {}, s1 = {};
        const _Float16* krow0 = Kh + (size_t)(b * T_SZ + kk + r) * DK_SZ;
        const _Float16* krow1 = krow0 + 16 * DK_SZ;
#pragma unroll
        for (int c = 0; c < 4; ++c) {
            s0 = wmma_f16(qa[c], load_b_frag(krow0 + 32 * c, hf), s0);
            s1 = wmma_f16(qa[c], load_b_frag(krow1 + 32 * c, hf), s1);
        }

        // ---- online softmax (row = i + 8*hf, cols striped across half-wave)
        float p0[8], p1[8];
#pragma unroll
        for (int i = 0; i < 8; ++i) {
            const float x0 = s0[i] * scale;
            const float x1 = s1[i] * scale;
            float tmax = half_reduce_max(fmaxf(x0, x1));
            const float nm   = fmaxf(mi[i], tmax);
            const float corr = __expf(mi[i] - nm);
            const float e0 = __expf(x0 - nm);
            const float e1 = __expf(x1 - nm);
            const float rs = half_reduce_sum(e0 + e1);
            li[i] = li[i] * corr + rs;
            mi[i] = nm;
            p0[i] = e0;
            p1[i] = e1;
#pragma unroll
            for (int j = 0; j < 8; ++j) o[j][i] *= corr;
        }

        // ---- relayout P (C layout) -> A layout through per-wave LDS ----
#pragma unroll
        for (int i = 0; i < 8; ++i) {
            pbuf[wv][i + 8 * hf][r]      = (_Float16)p0[i];
            pbuf[wv][i + 8 * hf][16 + r] = (_Float16)p1[i];
        }
        asm volatile("s_wait_dscnt 0" ::: "memory");
        Frag16 pf;
        const uint4* pp = (const uint4*)&pbuf[wv][r][0];
        pf.u2[0] = pp[2 * hf];       // K = [16h, 16h+8)
        pf.u2[1] = pp[2 * hf + 1];   // K = [16h+8, 16h+16)
        const v16h pa = pf.h;

        // ---- O += P @ V  (V^T rows are contiguous in keys) ----
        const size_t vcol = (size_t)(b * T_SZ + kk) + 16 * hf;
#pragma unroll
        for (int j = 0; j < 8; ++j) {
            const _Float16* vrow = Vt + (size_t)(j * 16 + r) * BT_SZ + vcol;
            Frag16 vf;
            const uint4* vp = (const uint4*)vrow;
            vf.u2[0] = vp[0];
            vf.u2[1] = vp[1];
            o[j] = wmma_f16(pa, vf.h, o[j]);
        }
    }

    // ---- epilogue: normalize and store f16 ----
#pragma unroll
    for (int i = 0; i < 8; ++i) {
        const float inv = 1.0f / li[i];
        const size_t rowoff =
            (size_t)(b * T_SZ + q0 + i + 8 * hf) * D_SZ + h * DK_SZ;
#pragma unroll
        for (int j = 0; j < 8; ++j) {
            AO[rowoff + j * 16 + r] = (_Float16)(o[j][i] * inv);
        }
    }
}

// ---------------------------------------------------------------------------
// Host launch
// ---------------------------------------------------------------------------
extern "C" void kernel_launch(void* const* d_in, const int* in_sizes, int n_in,
                              void* d_out, int out_size, void* d_ws, size_t ws_size,
                              hipStream_t stream) {
    const float* x  = (const float*)d_in[0];
    const float* Wq = (const float*)d_in[1];
    const float* bq = (const float*)d_in[2];
    const float* Wk = (const float*)d_in[3];
    const float* bk = (const float*)d_in[4];
    const float* Wv = (const float*)d_in[5];
    const float* bv = (const float*)d_in[6];
    const float* Wo = (const float*)d_in[7];
    const float* bo = (const float*)d_in[8];

    const size_t nX  = (size_t)BT_SZ * D_SZ;    // 8,388,608
    const size_t nWq = (size_t)D_SZ * D_SZ;     // 4,194,304
    const size_t nWk = (size_t)D_SZ * DK_SZ;    // 262,144
    const size_t nKV = (size_t)BT_SZ * DK_SZ;   // 524,288

    char* ws = (char*)d_ws;
    size_t off = 0;
    auto carve = [&](size_t elems) -> _Float16* {
        _Float16* p = (_Float16*)(ws + off);
        off += ((elems * sizeof(_Float16)) + 255) & ~(size_t)255;
        return p;
    };
    _Float16* xh  = carve(nX);
    _Float16* Wqt = carve(nWq);
    _Float16* Wkt = carve(nWk);
    _Float16* Wvt = carve(nWk);
    _Float16* Wot = carve(nWq);
    _Float16* Qhh = carve(nX);
    _Float16* Khh = carve(nKV);
    _Float16* Vth = carve(nKV);
    _Float16* AOh = carve(nX);

    // 1) conversions
    cvt_f32_to_f16<<<(int)(nX / 256), 256, 0, stream>>>(x, xh, (int)nX);
    cvt_transpose_f32_to_f16<<<(int)(nWq / 256), 256, 0, stream>>>(Wq, Wqt, D_SZ, D_SZ);
    cvt_transpose_f32_to_f16<<<(int)(nWk / 256), 256, 0, stream>>>(Wk, Wkt, D_SZ, DK_SZ);
    cvt_transpose_f32_to_f16<<<(int)(nWk / 256), 256, 0, stream>>>(Wv, Wvt, D_SZ, DK_SZ);
    cvt_transpose_f32_to_f16<<<(int)(nWq / 256), 256, 0, stream>>>(Wo, Wot, D_SZ, D_SZ);

    // 2) projections
    //    Q: (M/128)*(N/64) = 32*32 = 1024 blocks
    gemm_wmma_f16<false, false><<<1024, 256, 0, stream>>>(
        xh, Wqt, bq, (void*)Qhh, BT_SZ, D_SZ, D_SZ);
    //    K: 32*2 = 64 blocks
    gemm_wmma_f16<false, false><<<64, 256, 0, stream>>>(
        xh, Wkt, bk, (void*)Khh, BT_SZ, DK_SZ, D_SZ);
    //    V: transposed store -> Vt[128, 4096]
    gemm_wmma_f16<false, true><<<64, 256, 0, stream>>>(
        xh, Wvt, bv, (void*)Vth, BT_SZ, DK_SZ, D_SZ);

    // 3) flash attention: B*H*(T/64) = 1024 blocks of 4 waves
    flash_mqa<<<B_SZ * H_SZ * (T_SZ / 64), 128, 0, stream>>>(Qhh, Khh, Vth, AOh);

    // 4) output projection -> f32 d_out
    gemm_wmma_f16<true, false><<<1024, 256, 0, stream>>>(
        AOh, Wot, bo, d_out, BT_SZ, D_SZ, D_SZ);
}
